// Conv2dWeightModulateNoStyle_87282325389320
// MI455X (gfx1250) — matrix operand order repaired and checked by reading, not compiled, not run on hardware
//
#include <hip/hip_runtime.h>
#include <hip/hip_bf16.h>

// ---------------------------------------------------------------------------
// Conv2d (modulated/demodulated weights, no style) as implicit GEMM with
// CDNA5 WMMA bf16 16x16x32, fp32 emulated via bf16 hi/lo split (3 WMMAs).
// ---------------------------------------------------------------------------

typedef __attribute__((ext_vector_type(16))) __bf16 v16bf;
typedef __attribute__((ext_vector_type(8)))  __bf16 v8bf;
typedef __attribute__((ext_vector_type(8)))  float  v8f;

#define BATCH   16
#define CIN     64
#define COUT    64
#define HGT     256
#define WID     256
#define HW      (HGT * WID)
#define CHW     (CIN * HW)
#define TAPS    9
#define KTOT    (CIN * TAPS)        // 576
#define C_EQF   0.041666666666666664f  // 1/sqrt(64*9) = 1/24
#define EPSF    1e-5f

#define WTILE   64                  // pixels (along W) per block
#define HSTRIP  8                   // output rows per block
#define BUFW    (WTILE + 2)         // 66 : halo of 1 on each side
#define NTHREADS 512                // 16 waves (wave32)

// ---------------------------------------------------------------------------
// Kernel 1: demodulate weights, transpose [O][I][3][3] -> [co][tap][ci],
// split fp32 into bf16 hi + bf16 lo, write to workspace.
// ---------------------------------------------------------------------------
__global__ __launch_bounds__(CIN)
void prep_weights_kernel(const float* __restrict__ w,
                         __bf16* __restrict__ whi,
                         __bf16* __restrict__ wlo)
{
    const int co = blockIdx.x;      // 0..63
    const int ci = threadIdx.x;     // 0..63
    __shared__ float ps[CIN];

    const float* wc = w + (size_t)co * KTOT;   // [ci][tap], tap minor
    float s = 0.f;
    #pragma unroll
    for (int tap = 0; tap < TAPS; ++tap) {
        float v = wc[ci * TAPS + tap] * C_EQF;
        s += v * v;
    }
    ps[ci] = s;
    __syncthreads();
    // Every thread sums all 64 partials in the same order -> deterministic.
    float tot = 0.f;
    for (int i = 0; i < CIN; ++i) tot += ps[i];
    const float m = C_EQF * rsqrtf(tot + EPSF);

    #pragma unroll
    for (int tap = 0; tap < TAPS; ++tap) {
        float v = wc[ci * TAPS + tap] * m;
        __bf16 hi = (__bf16)v;
        float  lo = v - (float)hi;
        const int o = (co * TAPS + tap) * CIN + ci;
        whi[o] = hi;
        wlo[o] = (__bf16)lo;
    }
}

// ---------------------------------------------------------------------------
// Stage one input row (with W halo) into the LDS ring buffer, transposed to
// [px][ci] with bf16 hi/lo split. Out-of-range rows/pixels become zeros.
// ---------------------------------------------------------------------------
__device__ __forceinline__
void stage_row(const float* __restrict__ xb, int row, int w0,
               __bf16* __restrict__ shh, __bf16* __restrict__ shl, int tid)
{
    const int slot = (row + 1) % 3;            // row >= -1
    __bf16* dh = shh + slot * (BUFW * CIN);
    __bf16* dl = shl + slot * (BUFW * CIN);
    const bool rowok = (row >= 0) && (row < HGT);
    const float* xr = xb + (size_t)row * WID;

    for (int idx = tid; idx < BUFW * CIN; idx += NTHREADS) {
        const int ci = idx / BUFW;             // lanes walk consecutive px -> coalesced
        const int p  = idx - ci * BUFW;
        const int w  = w0 - 1 + p;
        float v = 0.f;
        if (rowok && w >= 0 && w < WID) v = xr[(size_t)ci * HW + w];
        __bf16 hi = (__bf16)v;
        float  lo = v - (float)hi;
        dh[p * CIN + ci] = hi;
        dl[p * CIN + ci] = (__bf16)lo;
    }
}

// ---------------------------------------------------------------------------
// Kernel 2: implicit-GEMM conv. Block = 16 waves covering 64 co x 64 px for
// an 8-row H strip; per-wave tile = 16 co x 16 px; K loop = 9 taps x 2 halves
// of ci, 3 bf16 WMMAs (hi*hi + hi*lo + lo*hi) per K=32 chunk.
// ---------------------------------------------------------------------------
__global__ __launch_bounds__(NTHREADS)
void conv_wmma_kernel(const float* __restrict__ x,
                      const __bf16* __restrict__ whi,
                      const __bf16* __restrict__ wlo,
                      float* __restrict__ out)
{
    __shared__ __align__(32) __bf16 sh_hi[3 * BUFW * CIN];
    __shared__ __align__(32) __bf16 sh_lo[3 * BUFW * CIN];

    const int wt = blockIdx.x;                 // 0..3   W tile
    const int hs = blockIdx.y;                 // 0..31  H strip
    const int b  = blockIdx.z;                 // 0..15  batch
    const int w0 = wt * WTILE;
    const int h0 = hs * HSTRIP;

    const int tid  = threadIdx.x;
    const int lane = tid & 31;
    const int wave = tid >> 5;                 // 0..15
    const int co_base = (wave & 3) * 16;
    const int px_base = (wave >> 2) * 16;
    const int n   = lane & 15;                 // N index (pixel) / A row select
    const int hf  = lane >> 4;                 // lane half: picks K sub-chunks

    const float* xb = x + (size_t)b * CHW;

    // Prime the 3-row ring buffer.
    stage_row(xb, h0 - 1, w0, sh_hi, sh_lo, tid);
    stage_row(xb, h0,     w0, sh_hi, sh_lo, tid);
    stage_row(xb, h0 + 1, w0, sh_hi, sh_lo, tid);
    __syncthreads();

    for (int it = 0; it < HSTRIP; ++it) {
        const int h = h0 + it;
        v8f acc = {};

        #pragma unroll 3
        for (int tap = 0; tap < TAPS; ++tap) {
            const int kh = tap / 3;
            const int kw = tap - kh * 3;
            const int slot = (h + kh) % 3;     // row (h + kh - 1) lives here
            const int p = px_base + n + kw;    // buffer px for this lane/tap
            const __bf16* srow_hi = sh_hi + (slot * BUFW + p) * CIN;
            const __bf16* srow_lo = sh_lo + (slot * BUFW + p) * CIN;
            const int widx = ((co_base + n) * TAPS + tap) * CIN;

            #pragma unroll
            for (int c0 = 0; c0 < CIN; c0 += 32) {
                // A fragment (weights): lanes 0-15 take K 0..15,
                // lanes 16-31 take K 16..31 of this 32-chunk.
                union { v16bf v; v8bf h2[2]; } ahi, alo;
                ahi.v = *(const v16bf*)(whi + widx + c0 + hf * 16);
                alo.v = *(const v16bf*)(wlo + widx + c0 + hf * 16);

                // B fragment (input): lanes 0-15 hold K=c0..c0+15 for pixel n,
                // lanes 16-31 hold K=c0+16..c0+31 -> 16 consecutive ci.
                v16bf bhi = *(const v16bf*)(srow_hi + c0 + hf * 16);
                v16bf blo = *(const v16bf*)(srow_lo + c0 + hf * 16);

                acc = __builtin_amdgcn_wmma_f32_16x16x32_bf16(
                        false, ahi.v, false, bhi, (short)0, acc, false, false);
                acc = __builtin_amdgcn_wmma_f32_16x16x32_bf16(
                        false, ahi.v, false, blo, (short)0, acc, false, false);
                acc = __builtin_amdgcn_wmma_f32_16x16x32_bf16(
                        false, alo.v, false, bhi, (short)0, acc, false, false);
            }
        }

        // D layout: VGPR r -> co = co_base + r + hf*8, N = pixel n.
        float* op = out + (size_t)b * ((size_t)COUT * HW)
                        + (size_t)(co_base + hf * 8) * HW
                        + (size_t)h * WID + (w0 + px_base + n);
        #pragma unroll
        for (int r = 0; r < 8; ++r) op[(size_t)r * HW] = acc[r];

        __syncthreads();                       // everyone done reading slot(h-1)
        if (it < HSTRIP - 1)
            stage_row(xb, h + 2, w0, sh_hi, sh_lo, tid);
        __syncthreads();                       // new row visible
    }
}

// ---------------------------------------------------------------------------
extern "C" void kernel_launch(void* const* d_in, const int* in_sizes, int n_in,
                              void* d_out, int out_size, void* d_ws, size_t ws_size,
                              hipStream_t stream) {
    const float* x = (const float*)d_in[0];   // [16,64,256,256]
    const float* w = (const float*)d_in[1];   // [64,64,3,3]
    float* out = (float*)d_out;               // [16,64,256,256]

    __bf16* whi = (__bf16*)d_ws;              // [64][9][64] bf16 hi
    __bf16* wlo = whi + (size_t)COUT * KTOT;  // [64][9][64] bf16 lo  (144 KB total)

    prep_weights_kernel<<<COUT, CIN, 0, stream>>>(w, whi, wlo);

    dim3 grid(WID / WTILE, HGT / HSTRIP, BATCH);  // (4, 32, 16)
    conv_wmma_kernel<<<grid, NTHREADS, 0, stream>>>(x, whi, wlo, out);
}
